// Seq2SeqLstmAE_58248346469069
// MI455X (gfx1250) — matrix operand order repaired and checked by reading, not compiled, not run on hardware
//
#include <hip/hip_runtime.h>

// Problem constants (from the reference)
#define BB 512
#define TB 1024
#define FF 32
#define HH 64
#define SS 16

typedef _Float16 half_t;
typedef __attribute__((ext_vector_type(16))) _Float16 v16h;
typedef __attribute__((ext_vector_type(8)))  float    v8f;

union V16H { v16h v; unsigned u[8]; half_t h[16]; };
union V8F  { v8f  v; float f[8]; };

__device__ __forceinline__ float ftanh_(float x) {
  float e = __expf(2.0f * x);
  return 1.0f - 2.0f * __builtin_amdgcn_rcpf(e + 1.0f);
}

// DPP quad broadcast: ctrl = 0x00/0x55/0xAA/0xFF broadcasts quad lane 0/1/2/3.
template <int CTRL>
__device__ __forceinline__ float quad_bcast(float x) {
  int xi = __builtin_bit_cast(int, x);
  int r  = __builtin_amdgcn_update_dpp(xi, xi, CTRL, 0xf, 0xf, true);
  return __builtin_bit_cast(float, r);
}

// A-fragment (16x32 f16, M=row) loaded from an LDS f16 matrix with row stride `stride`.
// Layout per ISA 7.12.2; contiguous per-lane pairs fuse into ds_load_b128.
__device__ __forceinline__ v16h load_a_frag(const half_t* __restrict__ buf, int stride, int kb, int lane) {
  const int M = lane & 15, hi = lane >> 4;
  const unsigned* p = (const unsigned*)buf;
  V16H r;
#pragma unroll
  for (int v = 0; v < 8; ++v) {
    int kl = ((v < 4) ? (2 * v) : (16 + 2 * (v - 4))) + (hi ? 8 : 0);
    r.u[v] = p[(M * stride + kb + kl) >> 1];
  }
  return r.v;
}

// B-fragment (32x16 f16, N=lane%16) built once from global f32 weights.
__device__ __forceinline__ v16h make_b_frag(const float* __restrict__ W, int ld, int row, int kbase, int lane) {
  const int hi = lane >> 4;
  V16H r;
#pragma unroll
  for (int v = 0; v < 8; ++v) {
    int k = kbase + ((v < 4) ? (2 * v) : (16 + 2 * (v - 4))) + (hi ? 8 : 0);
    const float* p = W + row * ld + k;
    r.h[2 * v]     = (half_t)p[0];
    r.h[2 * v + 1] = (half_t)p[1];
  }
  return r.v;
}

// LSTM cell: each lane activates ITS OWN gate first (1 exp + 1 rcp), then the
// activated values are quad-broadcast via DPP. Only quad-leader lanes keep valid
// c/h state (all consumers — LDS h store, output projection — read leaders only;
// the WMMA A operand always comes from LDS).
// ga = 2 for tanh lanes (gate g, lane&3==2), 1 for sigmoid lanes; gb = ga-1.
//   act = ga * rcp(1 + exp(-ga*x)) - gb   ->  sigmoid(x) or tanh(x)
__device__ __forceinline__ void lstm_cell(V8F& acc, float* creg, float* hreg,
                                          float ga, float gb,
                                          bool use_mask, int t, const int* len8) {
#pragma unroll
  for (int r = 0; r < 8; ++r) {
    float gv  = acc.f[r];
    float e   = __expf(-ga * gv);
    float rc  = __builtin_amdgcn_rcpf(1.0f + e);
    float act = __builtin_fmaf(ga, rc, -gb);
    float ai = quad_bcast<0x00>(act);
    float af = quad_bcast<0x55>(act);
    float ag = quad_bcast<0xAA>(act);
    float ao = quad_bcast<0xFF>(act);
    float cn = __builtin_fmaf(af, creg[r], ai * ag);
    float hn = ao * ftanh_(cn);
    if (use_mask) {
      bool on = (t < len8[r]);
      cn = on ? cn : creg[r];
      hn = on ? hn : hreg[r];
    }
    creg[r] = cn; hreg[r] = hn;
  }
}

__device__ __forceinline__ void store_h(half_t* __restrict__ hnext, const float* hreg,
                                        int lane, int hi, int unit) {
  if ((lane & 3) == 0) {
#pragma unroll
    for (int r = 0; r < 8; ++r) hnext[(r + 8 * hi) * 72 + unit] = (half_t)hreg[r];
  }
}

// Fused O=1 projection: quad-leader lanes ds_add_f32 directly.
__device__ __forceinline__ void out_partial(const float* hreg, float w_h, int lane, int hi,
                                            float* __restrict__ accslot) {
  if ((lane & 3) == 0) {
#pragma unroll
    for (int r = 0; r < 8; ++r) atomicAdd(&accslot[r + 8 * hi], hreg[r] * w_h);
  }
}

__global__ __launch_bounds__(512)
void seq2seq_lstm_ae_kernel(const float* __restrict__ to_x, const float* __restrict__ context,
                            const int* __restrict__ lengths,
                            const float* __restrict__ enc_Wih, const float* __restrict__ enc_Whh,
                            const float* __restrict__ enc_b,
                            const float* __restrict__ fc1_W, const float* __restrict__ fc1_b,
                            const float* __restrict__ dec_Wih, const float* __restrict__ dec_Whh,
                            const float* __restrict__ dec_b,
                            const float* __restrict__ out_W, const float* __restrict__ out_b,
                            float* __restrict__ out) {
  __shared__ __align__(16) half_t hbuf[2][16 * 72];   // double-buffered hidden state (f16, padded)
  __shared__ __align__(16) half_t xbuf[2][16 * 40];   // double-buffered x_t tile (f16)
  __shared__ __align__(16) half_t x0buf[16 * 72];     // fc1 output (first decoder input)
  __shared__ float out_acc[2][16];                    // fused output accumulator

  const int tid  = threadIdx.x;
  const int wave = tid >> 5;
  const int lane = tid & 31;
  const int hi   = lane >> 4;
  const int col  = lane & 15;
  const int wgbase = blockIdx.x * 16;

  // zero LDS
  for (int i = tid; i < 2 * 16 * 72; i += 512) hbuf[0][i] = (half_t)0.f;
  for (int i = tid; i < 2 * 16 * 40; i += 512) xbuf[0][i] = (half_t)0.f;
  for (int i = tid; i < 16 * 72;     i += 512) x0buf[i]   = (half_t)0.f;
  if (tid < 32) out_acc[tid >> 4][tid & 15] = 0.f;

  // gate-permuted column: p -> (gate = p&3, unit = p>>2); original row = gate*64 + unit
  const int p    = wave * 16 + col;
  const int rowp = (p & 3) * 64 + (p >> 2);
  const int unit = p >> 2;
  const float ga = ((lane & 3) == 2) ? 2.0f : 1.0f;   // tanh vs sigmoid lane
  const float gb = ga - 1.0f;

  int len8[8];
#pragma unroll
  for (int r = 0; r < 8; ++r) len8[r] = lengths[wgbase + r + 8 * hi];
  // lengths globally sorted descending -> chunk max is the first row; steps beyond
  // it are exact no-ops under the mask, so the encoder loop can stop there.
  const int tmax = lengths[wgbase];

  // encoder weight fragments (K = 32 from x, 64 from h)
  v16h efx  = make_b_frag(enc_Wih, FF, rowp, 0,  lane);
  v16h efh0 = make_b_frag(enc_Whh, HH, rowp, 0,  lane);
  v16h efh1 = make_b_frag(enc_Whh, HH, rowp, 32, lane);
  const float bias_e = enc_b[rowp];

  float creg[8], hreg[8];
#pragma unroll
  for (int r = 0; r < 8; ++r) { creg[r] = 0.f; hreg[r] = 0.f; }

  __syncthreads();
  int cur = 0;

  // ---------------- encoder ----------------
  for (int t = 0; t < tmax; ++t) {
    {   // stage x_t tile (16 rows x 32 feats) into LDS as f16; prefetch t+1
      const int r = tid >> 5, k = tid & 31;
      const float* xp = to_x + (size_t)(wgbase + r) * (TB * FF) + (size_t)t * FF + k;
      xbuf[t & 1][r * 40 + k] = (half_t)xp[0];
      if (t + 1 < tmax) __builtin_prefetch(xp + FF, 0, 1);
    }
    __syncthreads();

    // Loads issued in the same order WMMAs consume them -> partial s_wait_dscnt
    // lets outstanding ds_loads overlap WMMA execution.
    v16h a0 = load_a_frag(hbuf[cur],   72, 0,  lane);
    v16h a1 = load_a_frag(hbuf[cur],   72, 32, lane);
    v16h ax = load_a_frag(xbuf[t & 1], 40, 0,  lane);
    V8F acc0, acc1;
#pragma unroll
    for (int r = 0; r < 8; ++r) { acc0.f[r] = bias_e; acc1.f[r] = 0.f; }
    acc0.v = __builtin_amdgcn_wmma_f32_16x16x32_f16(false, a0, false, efh0, (short)0, acc0.v, false, false);
    acc1.v = __builtin_amdgcn_wmma_f32_16x16x32_f16(false, a1, false, efh1, (short)0, acc1.v, false, false);
    acc0.v = __builtin_amdgcn_wmma_f32_16x16x32_f16(false, ax, false, efx,  (short)0, acc0.v, false, false);
    V8F acc;
#pragma unroll
    for (int r = 0; r < 8; ++r) acc.f[r] = acc0.f[r] + acc1.f[r];

    lstm_cell(acc, creg, hreg, ga, gb, true, t, len8);
    store_h(hbuf[cur ^ 1], hreg, lane, hi, unit);   // frozen rows re-store old h
    cur ^= 1;
  }
  __syncthreads();   // h_enc visible in hbuf[cur]

  // ---------------- fc1: x0 = h_enc @ fc1_W.T + b (waves 0..3, N=64) ----------------
  if (wave < 4) {
    const int n = wave * 16 + col;
    v16h w0 = make_b_frag(fc1_W, HH, n, 0,  lane);
    v16h w1 = make_b_frag(fc1_W, HH, n, 32, lane);
    v16h a0 = load_a_frag(hbuf[cur], 72, 0,  lane);
    v16h a1 = load_a_frag(hbuf[cur], 72, 32, lane);
    V8F acc0, acc1;
    const float fb = fc1_b[n];
#pragma unroll
    for (int r = 0; r < 8; ++r) { acc0.f[r] = fb; acc1.f[r] = 0.f; }
    acc0.v = __builtin_amdgcn_wmma_f32_16x16x32_f16(false, a0, false, w0, (short)0, acc0.v, false, false);
    acc1.v = __builtin_amdgcn_wmma_f32_16x16x32_f16(false, a1, false, w1, (short)0, acc1.v, false, false);
#pragma unroll
    for (int r = 0; r < 8; ++r) x0buf[(r + 8 * hi) * 72 + n] = (half_t)(acc0.f[r] + acc1.f[r]);
  }

  // decoder weight fragments; for steps>=1, x_in==h so gates = h @ (Wih+Whh).T
  v16h dwi0 = make_b_frag(dec_Wih, HH, rowp, 0,  lane);
  v16h dwi1 = make_b_frag(dec_Wih, HH, rowp, 32, lane);
  v16h dwh0 = make_b_frag(dec_Whh, HH, rowp, 0,  lane);
  v16h dwh1 = make_b_frag(dec_Whh, HH, rowp, 32, lane);
  v16h dws0 = dwi0 + dwh0;   // v_pk_add_f16
  v16h dws1 = dwi1 + dwh1;
  const float bias_d = dec_b[rowp];
  const float w_h = out_W[unit];

  float ctx_dot = 0.f;
  if (wave == 0 && lane < 16) {
    for (int s = 0; s < SS; ++s) ctx_dot += context[(wgbase + lane) * SS + s] * out_W[HH + s];
    ctx_dot += out_b[0];
  }
  __syncthreads();

  // ---------------- decoder step 0: x0 @ Wih.T + h_enc @ Whh.T ----------------
  {
    v16h x0a = load_a_frag(x0buf,     72, 0,  lane);
    v16h x0b = load_a_frag(x0buf,     72, 32, lane);
    v16h h0  = load_a_frag(hbuf[cur], 72, 0,  lane);
    v16h h1  = load_a_frag(hbuf[cur], 72, 32, lane);
    V8F acc0, acc1;
#pragma unroll
    for (int r = 0; r < 8; ++r) { acc0.f[r] = bias_d; acc1.f[r] = 0.f; }
    acc0.v = __builtin_amdgcn_wmma_f32_16x16x32_f16(false, x0a, false, dwi0, (short)0, acc0.v, false, false);
    acc1.v = __builtin_amdgcn_wmma_f32_16x16x32_f16(false, x0b, false, dwi1, (short)0, acc1.v, false, false);
    acc0.v = __builtin_amdgcn_wmma_f32_16x16x32_f16(false, h0,  false, dwh0, (short)0, acc0.v, false, false);
    acc1.v = __builtin_amdgcn_wmma_f32_16x16x32_f16(false, h1,  false, dwh1, (short)0, acc1.v, false, false);
    V8F acc;
#pragma unroll
    for (int r = 0; r < 8; ++r) acc.f[r] = acc0.f[r] + acc1.f[r];

    lstm_cell(acc, creg, hreg, ga, gb, false, 0, len8);
    out_partial(hreg, w_h, lane, hi, out_acc[0]);
    store_h(hbuf[cur ^ 1], hreg, lane, hi, unit);
    cur ^= 1;
  }

  // ---------------- decoder steps 1..T-1 ----------------
  for (int t = 1; t < TB; ++t) {
    __syncthreads();
    if (wave == 0 && lane < 16) {   // drain previous step's fused output
      const int slot = (t - 1) & 1;
      out[(size_t)(wgbase + lane) * TB + (t - 1)] = out_acc[slot][lane] + ctx_dot;
      out_acc[slot][lane] = 0.f;
    }
    v16h h0 = load_a_frag(hbuf[cur], 72, 0,  lane);
    v16h h1 = load_a_frag(hbuf[cur], 72, 32, lane);
    V8F acc0, acc1;
#pragma unroll
    for (int r = 0; r < 8; ++r) { acc0.f[r] = bias_d; acc1.f[r] = 0.f; }
    acc0.v = __builtin_amdgcn_wmma_f32_16x16x32_f16(false, h0, false, dws0, (short)0, acc0.v, false, false);
    acc1.v = __builtin_amdgcn_wmma_f32_16x16x32_f16(false, h1, false, dws1, (short)0, acc1.v, false, false);
    V8F acc;
#pragma unroll
    for (int r = 0; r < 8; ++r) acc.f[r] = acc0.f[r] + acc1.f[r];

    lstm_cell(acc, creg, hreg, ga, gb, false, t, len8);
    out_partial(hreg, w_h, lane, hi, out_acc[t & 1]);
    store_h(hbuf[cur ^ 1], hreg, lane, hi, unit);
    cur ^= 1;
  }
  __syncthreads();
  if (wave == 0 && lane < 16) {
    out[(size_t)(wgbase + lane) * TB + (TB - 1)] = out_acc[(TB - 1) & 1][lane] + ctx_dot;
  }
}

extern "C" void kernel_launch(void* const* d_in, const int* in_sizes, int n_in,
                              void* d_out, int out_size, void* d_ws, size_t ws_size,
                              hipStream_t stream) {
  (void)in_sizes; (void)n_in; (void)out_size; (void)d_ws; (void)ws_size;
  const float* to_x    = (const float*)d_in[0];
  const float* context = (const float*)d_in[1];
  const int*   lengths = (const int*)d_in[2];
  const float* enc_Wih = (const float*)d_in[3];
  const float* enc_Whh = (const float*)d_in[4];
  const float* enc_b   = (const float*)d_in[5];
  const float* fc1_W   = (const float*)d_in[6];
  const float* fc1_b   = (const float*)d_in[7];
  const float* dec_Wih = (const float*)d_in[8];
  const float* dec_Whh = (const float*)d_in[9];
  const float* dec_b   = (const float*)d_in[10];
  const float* out_W   = (const float*)d_in[11];
  const float* out_b   = (const float*)d_in[12];
  float* out = (float*)d_out;

  hipLaunchKernelGGL(seq2seq_lstm_ae_kernel, dim3(BB / 16), dim3(512), 0, stream,
                     to_x, context, lengths, enc_Wih, enc_Whh, enc_b,
                     fc1_W, fc1_b, dec_Wih, dec_Whh, dec_b, out_W, out_b, out);
}
